// DMGAGRUcell_77592879169776
// MI455X (gfx1250) — matrix operand order repaired and checked by reading, not compiled, not run on hardware
//
#include <hip/hip_runtime.h>
#include <hip/hip_bf16.h>

// ---------------- problem constants ----------------
#define N_NODES 2048
#define BATCH   16
#define DIN     2
#define UNITS   64
#define FEAT    66          // DIN + UNITS
#define FP      80          // FEAT padded to 5 WMMA col-tiles
#define XSK     224         // 3*FEAT=198 padded to 7 K-steps of 32
#define NEDGE   (N_NODES*16)
#define KSTEP   32

typedef __attribute__((ext_vector_type(8)))  float  v8f;
typedef __attribute__((ext_vector_type(16))) __bf16 v16bf;

union Frag { v16bf v; unsigned int u[8]; };

__device__ __forceinline__ unsigned short f2bf(float f) {
    unsigned int x = __float_as_uint(f);
    x += 0x7FFFu + ((x >> 16) & 1u);          // round-to-nearest-even
    return (unsigned short)(x >> 16);
}
__device__ __forceinline__ float bf2f(unsigned short h) {
    return __uint_as_float(((unsigned int)h) << 16);
}

// ---------------------------------------------------------------------------
// Weight convert: reference stacks xs as (f,m) interleaved (row = f*3+m).
// Our xs layout is m-major (row = m*66+f). Permute + pad to K=224, bf16.
// ---------------------------------------------------------------------------
__global__ void convw_kernel(const float* __restrict__ W, unsigned short* __restrict__ Wp, int OD) {
    int idx = blockIdx.x * blockDim.x + threadIdx.x;
    if (idx >= XSK * OD) return;
    int k = idx / OD, c = idx % OD;
    unsigned short h = 0;
    if (k < 3 * FEAT) {
        int m = k / FEAT, f = k % FEAT;           // our row order
        h = f2bf(W[(size_t)(f * 3 + m) * OD + c]);// reference row order
    }
    Wp[idx] = h;
}

// ---------------------------------------------------------------------------
// Build concat features: xb (B,N,80) bf16 operand, xs[0:66], zero pads & x1.
// phase==1: state = r * hx with r = ru[...,:UNITS]
// ---------------------------------------------------------------------------
__global__ void build_x_kernel(const float* __restrict__ inputs, const float* __restrict__ hx,
                               const float* __restrict__ ru, int phase,
                               unsigned short* __restrict__ xb, unsigned short* __restrict__ xs,
                               float* __restrict__ x1) {
    int idx = blockIdx.x * blockDim.x + threadIdx.x;      // b*N + n
    if (idx >= BATCH * N_NODES) return;
    int b = idx / N_NODES, n = idx % N_NODES;
    unsigned short* xbp = xb + (size_t)idx * FP;
    unsigned short* xsp = xs + (size_t)idx * XSK;
    float*          x1p = x1 + (size_t)idx * FEAT;
    const float* hxp = hx + (size_t)b * (N_NODES * UNITS) + (size_t)n * UNITS;
    const float* rup = ru + (size_t)idx * (2 * UNITS);

    float vals[FEAT];
    vals[0] = inputs[(size_t)b * (N_NODES * DIN) + (size_t)n * DIN + 0];
    vals[1] = inputs[(size_t)b * (N_NODES * DIN) + (size_t)n * DIN + 1];
    #pragma unroll 8
    for (int f = 0; f < UNITS; f++) {
        float s = hxp[f];
        if (phase) s *= rup[f];
        vals[2 + f] = s;
    }
    for (int f = 0; f < FEAT; f++) {
        unsigned short h = f2bf(vals[f]);
        xbp[f] = h; xsp[f] = h; x1p[f] = 0.0f;
    }
    for (int f = FEAT; f < FP; f++) xbp[f] = 0;    // GEMM col pad
    for (int k = 212; k < XSK; k++) xsp[k] = 0;    // xs K pad (198..211 zeroed by gemm_adp)
}

// ---------------------------------------------------------------------------
// SpMM: x1[b, rows[e], :] += vals[e] * x[b, cols[e], :]   (fp32 atomics)
// ---------------------------------------------------------------------------
__global__ void spmm_kernel(const int* __restrict__ srows, const int* __restrict__ scols,
                            const float* __restrict__ svals, const unsigned short* __restrict__ xb,
                            float* __restrict__ x1) {
    int idx = blockIdx.x * blockDim.x + threadIdx.x;      // (e, b)
    if (idx >= NEDGE * BATCH) return;
    int e = idx >> 4, b = idx & 15;
    int r = srows[e], c = scols[e];
    float v = svals[e];
    const unsigned short* src = xb + ((size_t)b * N_NODES + c) * FP;
    float* dst = x1 + ((size_t)b * N_NODES + r) * FEAT;
    #pragma unroll 3
    for (int f2 = 0; f2 < FEAT / 2; f2++) {
        unsigned int u = *(const unsigned int*)(src + 2 * f2);
        atomicAdd(dst + 2 * f2,     v * bf2f((unsigned short)(u & 0xFFFFu)));
        atomicAdd(dst + 2 * f2 + 1, v * bf2f((unsigned short)(u >> 16)));
    }
}

__global__ void pack_x1_kernel(const float* __restrict__ x1, unsigned short* __restrict__ xs) {
    int idx = blockIdx.x * blockDim.x + threadIdx.x;
    if (idx >= BATCH * N_NODES * FEAT) return;
    int bn = idx / FEAT, f = idx % FEAT;
    xs[(size_t)bn * XSK + FEAT + f] = f2bf(x1[idx]);
}

// ---------------------------------------------------------------------------
// x2 = adp[b] @ x[b]:  (2048x2048 fp32) @ (2048x80 bf16) -> xs[...,132:212] bf16
// WG = 4 waves, 64 rows x 80 cols. Double-buffered LDS, software-pipelined:
// fetch(k+32) overlaps WMMA(k); one barrier per K-step.
// ---------------------------------------------------------------------------
__global__ __launch_bounds__(128) void gemm_adp_kernel(const float* __restrict__ adp,
                                                       const unsigned short* __restrict__ xb,
                                                       unsigned short* __restrict__ xs) {
    __shared__ unsigned short ldsA[2][64 * KSTEP];   // [row][k]
    __shared__ unsigned short ldsB[2][FP * KSTEP];   // [col][k]
    const int b = blockIdx.y;
    const int rowBase = blockIdx.x * 64;
    const int t = threadIdx.x, wave = t >> 5, lane = t & 31;
    const float* Abase = adp + ((size_t)b * N_NODES + rowBase) * N_NODES;
    const unsigned short* Bbase = xb + (size_t)b * N_NODES * FP;
    const int rA = t >> 1, kcA = (t & 1) << 4;

    v8f acc[5];
    #pragma unroll
    for (int i = 0; i < 5; i++) acc[i] = (v8f){0,0,0,0,0,0,0,0};

    float4 fa[4];            // in-flight A tile (16 fp32)
    unsigned int bu[10];     // in-flight B tile (20 bf16)

    auto fetch = [&](int k0) {
        const float4* src = (const float4*)(Abase + (size_t)rA * N_NODES + k0 + kcA);
        #pragma unroll
        for (int q = 0; q < 4; q++) fa[q] = src[q];
        #pragma unroll
        for (int j = 0; j < 10; j++) {
            int i = t + 128 * j;
            int k = i / (FP / 2), f2i = i % (FP / 2);
            bu[j] = *(const unsigned int*)(Bbase + (size_t)(k0 + k) * FP + 2 * f2i);
        }
    };
    auto stage = [&](int p) {
        unsigned short* dst = &ldsA[p][rA * KSTEP + kcA];
        #pragma unroll
        for (int q = 0; q < 4; q++) {
            dst[4*q+0] = f2bf(fa[q].x); dst[4*q+1] = f2bf(fa[q].y);
            dst[4*q+2] = f2bf(fa[q].z); dst[4*q+3] = f2bf(fa[q].w);
        }
        #pragma unroll
        for (int j = 0; j < 10; j++) {
            int i = t + 128 * j;
            int k = i / (FP / 2), f2i = i % (FP / 2);
            ldsB[p][(2*f2i)     * KSTEP + k] = (unsigned short)(bu[j] & 0xFFFFu);
            ldsB[p][(2*f2i + 1) * KSTEP + k] = (unsigned short)(bu[j] >> 16);
        }
    };
    auto compute = [&](int p) {
        Frag a;
        const unsigned short* ap = &ldsA[p][(wave * 16 + (lane & 15)) * KSTEP];
        const int kh = (lane >> 4) ? 16 : 0;
        #pragma unroll
        for (int v = 0; v < 8; v++)
            a.u[v] = *(const unsigned int*)(ap + 2 * v + kh);
        #pragma unroll
        for (int ct = 0; ct < 5; ct++) {
            Frag bf;
            const unsigned short* bp = &ldsB[p][(ct * 16 + (lane & 15)) * KSTEP + ((lane >> 4) ? 16 : 0)];
            #pragma unroll
            for (int v = 0; v < 8; v++) bf.u[v] = *(const unsigned int*)(bp + 2 * v);
            acc[ct] = __builtin_amdgcn_wmma_f32_16x16x32_bf16(false, a.v, false, bf.v,
                                                              (short)0, acc[ct], false, false);
        }
    };

    fetch(0);
    stage(0);
    int p = 0;
    for (int k0 = KSTEP; k0 < N_NODES; k0 += KSTEP) {
        __syncthreads();     // buf p visible to all; buf p^1 free to overwrite
        fetch(k0);           // global loads in flight while we compute
        compute(p);
        stage(p ^ 1);
        p ^= 1;
    }
    __syncthreads();
    compute(p);

    // epilogue: cols 66..79 are exact zeros (B pad) -> xs[198:212] correctly zeroed
    const int coll = lane & 15, rsub = (lane >> 4) ? 8 : 0;
    #pragma unroll
    for (int ct = 0; ct < 5; ct++)
        #pragma unroll
        for (int j = 0; j < 8; j++) {
            int row = rowBase + wave * 16 + rsub + j;
            xs[((size_t)b * N_NODES + row) * XSK + 132 + ct * 16 + coll] = f2bf(acc[ct][j]);
        }
}

// ---------------------------------------------------------------------------
// out = act( xs(32768 x 224) @ Wp(224 x OD) ); whole Wp lives in LDS [col][k].
// A tile staged with CDNA5 async LDS copies (global_load_async_to_lds_b128).
// ACT 0 = sigmoid (ru), ACT 1 = tanh (c)
// ---------------------------------------------------------------------------
template <int OD, int ACT>
__global__ __launch_bounds__(128) void gemm_out_kernel(const unsigned short* __restrict__ xs,
                                                       const unsigned short* __restrict__ Wp,
                                                       float* __restrict__ out) {
    extern __shared__ unsigned short smem_dyn[];
    unsigned short* ldsA = smem_dyn;              // 64*32   [row][k]
    unsigned short* ldsB = smem_dyn + 64 * KSTEP; // OD*224  [col][k]
    const int t = threadIdx.x, wave = t >> 5, lane = t & 31;
    const int rowBase = blockIdx.x * 64;

    for (int i = t; i < XSK * OD; i += 128) {
        int k = i / OD, c = i % OD;
        ldsB[c * XSK + k] = Wp[i];
    }
    constexpr int NCT = OD / 16;
    v8f acc[NCT];
    #pragma unroll
    for (int i = 0; i < NCT; i++) acc[i] = (v8f){0,0,0,0,0,0,0,0};
    __syncthreads();

    for (int k0 = 0; k0 < XSK; k0 += KSTEP) {
        // async stage A tile: 64 rows x 64B, 256 x B128 chunks, 2 per thread.
        // LDS dest = low 32 bits of the flat shared address (LDS aperture keeps
        // the byte offset in addr[31:0]).
        #pragma unroll
        for (int c = 0; c < 2; c++) {
            int chunk = t + 128 * c;
            int r = chunk >> 2, q = chunk & 3;
            const unsigned short* gp = xs + (size_t)(rowBase + r) * XSK + k0 + q * 8;
            unsigned long long ga = (unsigned long long)(uintptr_t)gp;
            unsigned la = (unsigned)(uintptr_t)(ldsA + r * KSTEP + q * 8);
            asm volatile("global_load_async_to_lds_b128 %0, %1, off"
                         :: "v"(la), "v"(ga) : "memory");
        }
        asm volatile("s_wait_asynccnt 0x0" ::: "memory");
        __syncthreads();

        Frag a;
        {
            const unsigned short* ap = ldsA + (wave * 16 + (lane & 15)) * KSTEP;
            const int kh = (lane >> 4) ? 16 : 0;
            #pragma unroll
            for (int v = 0; v < 8; v++)
                a.u[v] = *(const unsigned int*)(ap + 2 * v + kh);
        }
        #pragma unroll
        for (int ct = 0; ct < NCT; ct++) {
            Frag bf;
            const unsigned short* bp = ldsB + (ct * 16 + (lane & 15)) * XSK + k0 + ((lane >> 4) ? 16 : 0);
            #pragma unroll
            for (int v = 0; v < 8; v++) bf.u[v] = *(const unsigned int*)(bp + 2 * v);
            acc[ct] = __builtin_amdgcn_wmma_f32_16x16x32_bf16(false, a.v, false, bf.v,
                                                              (short)0, acc[ct], false, false);
        }
        __syncthreads();
    }
    const int coll = lane & 15, rsub = (lane >> 4) ? 8 : 0;
    #pragma unroll
    for (int ct = 0; ct < NCT; ct++)
        #pragma unroll
        for (int j = 0; j < 8; j++) {
            int row = rowBase + wave * 16 + rsub + j;
            float x = acc[ct][j], y;
            if (ACT == 0) { y = 1.0f / (1.0f + __expf(-x)); }
            else          { float e = __expf(2.0f * x); y = (e - 1.0f) / (e + 1.0f); }
            out[(size_t)row * OD + ct * 16 + coll] = y;
        }
}

// ---------------------------------------------------------------------------
// GRU combine: out = u*hx + (1-u)*c
// ---------------------------------------------------------------------------
__global__ void final_kernel(const float* __restrict__ ru, const float* __restrict__ hx,
                             const float* __restrict__ c, float* __restrict__ out) {
    int idx = blockIdx.x * blockDim.x + threadIdx.x;
    if (idx >= BATCH * N_NODES * UNITS) return;
    int bn = idx / UNITS, f = idx % UNITS;
    float u = ru[(size_t)bn * (2 * UNITS) + UNITS + f];
    out[idx] = u * hx[idx] + (1.0f - u) * c[idx];
}

// ---------------------------------------------------------------------------
extern "C" void kernel_launch(void* const* d_in, const int* in_sizes, int n_in,
                              void* d_out, int out_size, void* d_ws, size_t ws_size,
                              hipStream_t stream) {
    const float* inputs = (const float*)d_in[0];
    const float* hx     = (const float*)d_in[1];
    const float* adp    = (const float*)d_in[2];
    const int*   srows  = (const int*)  d_in[3];
    const int*   scols  = (const int*)  d_in[4];
    const float* svals  = (const float*)d_in[5];
    const float* W_ru   = (const float*)d_in[6];
    const float* W_c    = (const float*)d_in[7];
    float* out = (float*)d_out;

    char* ws = (char*)d_ws;
    size_t off = 0;
    auto alloc = [&](size_t bytes) -> void* {
        void* p = ws + off; off += (bytes + 255) & ~(size_t)255; return p;
    };
    unsigned short* xb  = (unsigned short*)alloc((size_t)BATCH * N_NODES * FP  * 2);
    unsigned short* xs  = (unsigned short*)alloc((size_t)BATCH * N_NODES * XSK * 2);
    float*          x1  = (float*)         alloc((size_t)BATCH * N_NODES * FEAT * 4);
    float*          ru  = (float*)         alloc((size_t)BATCH * N_NODES * 2 * UNITS * 4);
    float*          cb  = (float*)         alloc((size_t)BATCH * N_NODES * UNITS * 4);
    unsigned short* Wru = (unsigned short*)alloc((size_t)XSK * 128 * 2);
    unsigned short* Wc  = (unsigned short*)alloc((size_t)XSK * 64  * 2);

    const int BN = BATCH * N_NODES;
    dim3 gAdp(N_NODES / 64, BATCH);

    convw_kernel<<<(XSK * 128 + 255) / 256, 256, 0, stream>>>(W_ru, Wru, 2 * UNITS);
    convw_kernel<<<(XSK * 64  + 255) / 256, 256, 0, stream>>>(W_c,  Wc,  UNITS);

    // ---- gconv 1: ru = sigmoid(gconv(inputs, hx) @ W_ru) ----
    build_x_kernel<<<(BN + 255) / 256, 256, 0, stream>>>(inputs, hx, ru, 0, xb, xs, x1);
    spmm_kernel<<<(NEDGE * BATCH + 255) / 256, 256, 0, stream>>>(srows, scols, svals, xb, x1);
    pack_x1_kernel<<<(BN * FEAT + 255) / 256, 256, 0, stream>>>(x1, xs);
    gemm_adp_kernel<<<gAdp, 128, 0, stream>>>(adp, xb, xs);
    gemm_out_kernel<128, 0><<<BN / 64, 128, (64 * KSTEP + XSK * 128) * 2, stream>>>(xs, Wru, ru);

    // ---- gconv 2: c = tanh(gconv(inputs, r*hx) @ W_c) ----
    build_x_kernel<<<(BN + 255) / 256, 256, 0, stream>>>(inputs, hx, ru, 1, xb, xs, x1);
    spmm_kernel<<<(NEDGE * BATCH + 255) / 256, 256, 0, stream>>>(srows, scols, svals, xb, x1);
    pack_x1_kernel<<<(BN * FEAT + 255) / 256, 256, 0, stream>>>(x1, xs);
    gemm_adp_kernel<<<gAdp, 128, 0, stream>>>(adp, xb, xs);
    gemm_out_kernel<64, 1><<<BN / 64, 128, (64 * KSTEP + XSK * 64) * 2, stream>>>(xs, Wc, cb);

    // ---- out = u*hx + (1-u)*c ----
    final_kernel<<<(BN * UNITS + 255) / 256, 256, 0, stream>>>(ru, hx, cb, out);

    (void)in_sizes; (void)n_in; (void)out_size; (void)ws_size;
}
